// ASRxLSTM_7327214207713
// MI455X (gfx1250) — matrix operand order repaired and checked by reading, not compiled, not run on hardware
//
#include <hip/hip_runtime.h>
#include <hip/hip_bf16.h>
#include <math.h>

// ---------------------------------------------------------------------------
// ASR xLSTM forward for MI455X (gfx1250).
// Big GEMMs use v_wmma_f32_16x16x32_bf16 (f32 accumulate, bf16 inputs staged
// through LDS as packed k-pairs). All GEMM shapes in this model are exact
// multiples of the 64x64 block tile / 32 K-step, so the staging path is
// branch-free with float4 (b128) global loads.
// ---------------------------------------------------------------------------

#define BB   4
#define TT   1024
#define FF   80
#define DD   512
#define DII  1024
#define NHH  4
#define DHM  256
#define DHS  128
#define NBB  256
#define DUPP 704
#define CC   1024
#define BT   4096          // B*T
#define EPSF 1e-5f

typedef __attribute__((ext_vector_type(16))) __bf16 v16bf;
typedef __attribute__((ext_vector_type(8)))  float  v8f;

union FragBF { v16bf v; unsigned u[8]; };

__device__ __forceinline__ unsigned pack_bf16(float a, float b)
{
    union { __bf16 h[2]; unsigned u; } t;
    t.h[0] = (__bf16)a;
    t.h[1] = (__bf16)b;
    return t.u;                 // lowers to v_cvt_pk_bf16_f32
}

// ======================= generic WMMA GEMM =================================
// C[M,N] = act( alpha * A[M,K] @ W + bias )   (optionally +=)
// opB==0: W is (N,K) row-major (x @ w.T case).  opB==1: W is (K,N) row-major.
// Batched over grid.z: z -> (b = z/NHz, h = z%NHz); base += b*s?b + h*s?h.
// Block: 256 threads = 8 waves in 2(M)x4(N); each wave owns two M-subtiles
// -> 64x64 block tile, K step 32, 2 WMMAs per wave per stage.
// REQUIRES: M%64==0, N%64==0, K%32==0, lda/ldw multiples of 4, 16B-aligned.
__global__ void gemm_bf16_wmma(const float* __restrict__ A,
                               const float* __restrict__ W,
                               const float* __restrict__ bias,
                               float* __restrict__ C,
                               int M, int N, int K,
                               int lda, int ldw, int ldc,
                               int NHz,
                               long long sAb, long long sAh,
                               long long sWb, long long sWh,
                               long long sCb, long long sCh,
                               int opB, float alpha, int act, int accum)
{
    int z = blockIdx.z;
    int bb = z / NHz, hh = z % NHz;
    A += (long long)bb * sAb + (long long)hh * sAh;
    W += (long long)bb * sWb + (long long)hh * sWh;
    C += (long long)bb * sCb + (long long)hh * sCh;

    const int n0 = blockIdx.x * 64;
    const int m0 = blockIdx.y * 64;
    const int tid = threadIdx.x;

    // LDS tiles: packed bf16 k-pairs, one dword per (k/2, x).
    __shared__ unsigned As[64][17];   // [m][kpair], +1 pad
    __shared__ unsigned Bs[16][68];   // [kpair][n], +4 pad

    v8f acc0 = {0.f,0.f,0.f,0.f,0.f,0.f,0.f,0.f};
    v8f acc1 = {0.f,0.f,0.f,0.f,0.f,0.f,0.f,0.f};

    const int wid  = tid >> 5;        // 0..7
    const int lane = tid & 31;
    const int wm   = wid >> 2;        // 0..1 -> M subrange [wm*32, wm*32+32)
    const int wn   = wid & 3;         // 0..3 (N tile)
    const int half = lane >> 4;       // 0/1
    const int lr   = lane & 15;

    for (int k0 = 0; k0 < K; k0 += 32) {
        // ---- stage A tile (64 M x 32 K): 8 floats per thread, b128 loads
        {
            int m  = tid >> 2;
            int kq = (tid & 3) * 8;
            const float4* ap =
                (const float4*)(A + (long long)(m0 + m) * lda + k0 + kq);
            float4 a0 = ap[0], a1 = ap[1];
            int kp = kq >> 1;
            As[m][kp + 0] = pack_bf16(a0.x, a0.y);
            As[m][kp + 1] = pack_bf16(a0.z, a0.w);
            As[m][kp + 2] = pack_bf16(a1.x, a1.y);
            As[m][kp + 3] = pack_bf16(a1.z, a1.w);
        }
        // ---- stage B tile (32 K x 64 N)
        if (opB == 0) {
            int n  = tid >> 2;
            int kq = (tid & 3) * 8;
            const float4* wp =
                (const float4*)(W + (long long)(n0 + n) * ldw + k0 + kq);
            float4 b0 = wp[0], b1 = wp[1];
            int kp = kq >> 1;
            Bs[kp + 0][n] = pack_bf16(b0.x, b0.y);
            Bs[kp + 1][n] = pack_bf16(b0.z, b0.w);
            Bs[kp + 2][n] = pack_bf16(b1.x, b1.y);
            Bs[kp + 3][n] = pack_bf16(b1.z, b1.w);
        } else {
            int p  = tid >> 4;            // k-pair 0..15
            int nq = (tid & 15) * 4;      // n 0..60
            const float4* w0 =
                (const float4*)(W + (long long)(k0 + 2 * p) * ldw + n0 + nq);
            const float4* w1 =
                (const float4*)(W + (long long)(k0 + 2 * p + 1) * ldw + n0 + nq);
            float4 r0 = *w0, r1 = *w1;
            Bs[p][nq + 0] = pack_bf16(r0.x, r1.x);
            Bs[p][nq + 1] = pack_bf16(r0.y, r1.y);
            Bs[p][nq + 2] = pack_bf16(r0.z, r1.z);
            Bs[p][nq + 3] = pack_bf16(r0.w, r1.w);
        }
        __syncthreads();

        // ---- fragments per ISA layout (CDNA5 ISA §7.12.2)
        // A frag dword i -> k-pair ((i<4)?0:8) + half*4 + (i&3)
        // B frag dword i -> k-pair ((i<4)?0:8) + half*4 + (i&3), column = lane&15
        FragBF fa0, fa1, fb;
        int mrow0 = m0 ? 0 : 0; (void)mrow0;
        int ra = wm * 32 + lr;            // acc0 rows
        int rb = ra + 16;                 // acc1 rows
        int ncol = wn * 16 + lr;
        #pragma unroll
        for (int i = 0; i < 8; ++i) {
            int kidx = ((i < 4) ? 0 : 8) + half * 4 + (i & 3);
            fa0.u[i] = As[ra][kidx];
            fa1.u[i] = As[rb][kidx];
            fb.u[i]  = Bs[kidx][ncol];
        }

#if __has_builtin(__builtin_amdgcn_wmma_f32_16x16x32_bf16)
        acc0 = __builtin_amdgcn_wmma_f32_16x16x32_bf16(
                   false, fa0.v, false, fb.v, (short)0, acc0, false, false);
        acc1 = __builtin_amdgcn_wmma_f32_16x16x32_bf16(
                   false, fa1.v, false, fb.v, (short)0, acc1, false, false);
#else
        // scalar fallback (keeps file compiling if builtin is absent)
        #pragma unroll
        for (int i = 0; i < 8; ++i) {
            int mm = wm * 32 + half * 8 + i;
            int nn = wn * 16 + lr;
            float s0 = 0.f, s1 = 0.f;
            for (int kk = 0; kk < 32; ++kk) {
                union { unsigned u; __bf16 h[2]; } ba, bbv, ba1;
                ba.u  = As[mm][kk >> 1];
                ba1.u = As[mm + 16][kk >> 1];
                bbv.u = Bs[kk >> 1][nn];
                s0 += (float)ba.h[kk & 1]  * (float)bbv.h[kk & 1];
                s1 += (float)ba1.h[kk & 1] * (float)bbv.h[kk & 1];
            }
            acc0[i] += s0; acc1[i] += s1;
        }
#endif
        __syncthreads();
    }

    // ---- epilogue: C dword i -> (M = half*8 + i, N = lane&15)
    #pragma unroll
    for (int i = 0; i < 8; ++i) {
        int gm0 = m0 + wm * 32 + half * 8 + i;
        int gn  = n0 + wn * 16 + lr;
        #pragma unroll
        for (int s = 0; s < 2; ++s) {
            int gm = gm0 + s * 16;
            float v = (s == 0 ? acc0[i] : acc1[i]) * alpha;
            if (bias) v += bias[gn];
            if (act == 1)      v = fmaxf(v, 0.f);
            else if (act == 2) v = v / (1.f + expf(-v));                 // silu
            else if (act == 3) v = 0.5f * v * (1.f + erff(v * 0.70710678118f));
            long long idx = (long long)gm * ldc + gn;
            if (accum) C[idx] += v; else C[idx] = v;
        }
    }
}

// ======================= front-end conv + BN + ReLU ========================
__global__ void conv_bn_relu(const float* __restrict__ x, int Cin, int Cout,
                             const float* __restrict__ w, const float* __restrict__ b,
                             const float* __restrict__ bng, const float* __restrict__ bnb,
                             const float* __restrict__ bnm, const float* __restrict__ bnv,
                             float* __restrict__ y, float inscale, int total)
{
    int idx = blockIdx.x * blockDim.x + threadIdx.x;
    if (idx >= total) return;
    int co = idx % Cout;
    int t  = (idx / Cout) % TT;
    int bb = idx / (Cout * TT);
    const float* wr = w + (long long)co * Cin * 3;
    float acc = 0.f;
    #pragma unroll
    for (int j = 0; j < 3; ++j) {
        int tt = t - 1 + j;
        if (tt < 0 || tt >= TT) continue;
        const float* xr = x + ((long long)bb * TT + tt) * Cin;
        for (int ci = 0; ci < Cin; ++ci) acc += xr[ci] * wr[ci * 3 + j];
    }
    acc = acc * inscale + b[co];
    acc = (acc - bnm[co]) * rsqrtf(bnv[co] + EPSF) * bng[co] + bnb[co];
    y[((long long)bb * TT + t) * Cout + co] = fmaxf(acc, 0.f);
}

// ======================= layernorm variants ================================
__device__ __forceinline__ float blk_reduce_sum(float v, float* sh, int nthr)
{
    int tid = threadIdx.x;
    sh[tid] = v; __syncthreads();
    for (int s = nthr >> 1; s > 0; s >>= 1) {
        if (tid < s) sh[tid] += sh[tid + s];
        __syncthreads();
    }
    float r = sh[0]; __syncthreads();
    return r;
}
__device__ __forceinline__ float blk_reduce_max(float v, float* sh, int nthr)
{
    int tid = threadIdx.x;
    sh[tid] = v; __syncthreads();
    for (int s = nthr >> 1; s > 0; s >>= 1) {
        if (tid < s) sh[tid] = fmaxf(sh[tid], sh[tid + s]);
        __syncthreads();
    }
    float r = sh[0]; __syncthreads();
    return r;
}

// one 256-thread block per row
__global__ void ln_kernel(const float* __restrict__ x, const float* __restrict__ w,
                          const float* __restrict__ b, float* __restrict__ y,
                          int n, int relu)
{
    __shared__ float sh[256];
    long long row = blockIdx.x;
    const float* xr = x + row * n;
    float* yr = y + row * n;
    int tid = threadIdx.x;
    float s = 0.f;
    for (int i = tid; i < n; i += 256) s += xr[i];
    float mu = blk_reduce_sum(s, sh, 256) / n;
    float s2 = 0.f;
    for (int i = tid; i < n; i += 256) { float d = xr[i] - mu; s2 += d * d; }
    float rstd = rsqrtf(blk_reduce_sum(s2, sh, 256) / n + EPSF);
    for (int i = tid; i < n; i += 256) {
        float v = (xr[i] - mu) * rstd * w[i];
        if (b) v += b[i];
        if (relu) v = fmaxf(v, 0.f);
        yr[i] = v;
    }
}

// mLSTM multi-head LN: in (B,NH,S,DHM) rows -> out (B,S,NH*DHM)
__global__ void mh_ln_kernel(const float* __restrict__ x, const float* __restrict__ w,
                             float* __restrict__ y)
{
    __shared__ float sh[256];
    long long r = blockIdx.x;                 // b*NH*S + h*S + s
    int s  = (int)(r % TT);
    int h  = (int)((r / TT) % NHH);
    int bb = (int)(r / ((long long)TT * NHH));
    const float* xr = x + r * DHM;
    const float* wr = w + h * DHM;
    float* yr = y + ((long long)bb * TT + s) * DII + h * DHM;
    int tid = threadIdx.x;
    float sm = 0.f;
    for (int i = tid; i < DHM; i += 256) sm += xr[i];
    float mu = blk_reduce_sum(sm, sh, 256) / DHM;
    float s2 = 0.f;
    for (int i = tid; i < DHM; i += 256) { float d = xr[i] - mu; s2 += d * d; }
    float rstd = rsqrtf(blk_reduce_sum(s2, sh, 256) / DHM + EPSF);
    for (int i = tid; i < DHM; i += 256)
        yr[i] = (xr[i] - mu) * rstd * wr[i];
}

// sLSTM group norm: rows of DHS, layout (B,S,NH,DHS)
__global__ void ln_grouped_kernel(const float* __restrict__ x, const float* __restrict__ w,
                                  float* __restrict__ y)
{
    __shared__ float sh[128];
    long long r = blockIdx.x;                 // b*S*NH + s*NH + h
    int h = (int)(r % NHH);
    const float* xr = x + r * DHS;
    const float* wr = w + h * DHS;
    float* yr = y + r * DHS;
    int tid = threadIdx.x;                    // 128 threads
    float v = xr[tid];
    float mu = blk_reduce_sum(v, sh, 128) / DHS;
    float d = v - mu;
    float rstd = rsqrtf(blk_reduce_sum(d * d, sh, 128) / DHS + EPSF);
    yr[tid] = d * rstd * wr[tid];
}

// ======================= elementwise helpers ===============================
__global__ void cconv_silu(const float* __restrict__ x, int ldx,
                           const float* __restrict__ w, const float* __restrict__ b,
                           float* __restrict__ y, int ldy, int Ch, int total)
{
    int idx = blockIdx.x * blockDim.x + threadIdx.x;
    if (idx >= total) return;
    int c  = idx % Ch;
    int t  = (idx / Ch) % TT;
    int bb = idx / (Ch * TT);
    float acc = b[c];
    #pragma unroll
    for (int j = 0; j < 4; ++j) {
        int tt = t - 3 + j;
        if (tt >= 0) acc += x[((long long)bb * TT + tt) * ldx + c] * w[c * 4 + j];
    }
    y[((long long)bb * TT + t) * ldy + c] = acc / (1.f + expf(-acc));
}

__global__ void headwise4(const float* __restrict__ x, int ldx,
                          const float* __restrict__ w,
                          float* __restrict__ y, int ldy, int total)
{
    int idx = blockIdx.x * blockDim.x + threadIdx.x;   // over BT*NBB
    if (idx >= total) return;
    int g  = idx % NBB;
    long long bt = idx / NBB;
    const float* xr = x + bt * ldx + g * 4;
    const float* wg = w + g * 16;
    float x0 = xr[0], x1 = xr[1], x2 = xr[2], x3 = xr[3];
    float* yr = y + bt * ldy + g * 4;
    #pragma unroll
    for (int o = 0; o < 4; ++o)
        yr[o] = x0 * wg[o * 4 + 0] + x1 * wg[o * 4 + 1] + x2 * wg[o * 4 + 2] + x3 * wg[o * 4 + 3];
}

// ig/fg gate projections: one block per token, 8 dots of length 3*Di
__global__ void mlstm_gates(const float* __restrict__ q, const float* __restrict__ k,
                            const float* __restrict__ v,
                            const float* __restrict__ igw, const float* __restrict__ igb,
                            const float* __restrict__ fgw, const float* __restrict__ fgb,
                            float* __restrict__ ig, float* __restrict__ fg)
{
    __shared__ float sh[256];
    long long bt = blockIdx.x;
    int b = (int)(bt / TT), t = (int)(bt % TT);
    int tid = threadIdx.x;
    float part[8] = {0,0,0,0,0,0,0,0};
    for (int i = tid; i < 3 * DII; i += 256) {
        float g;
        if (i < DII)            g = q[bt * DII + i];
        else if (i < 2 * DII)   g = k[bt * DII + i - DII];
        else                    g = v[bt * DII + i - 2 * DII];
        #pragma unroll
        for (int h = 0; h < NHH; ++h) {
            part[h]     += g * igw[h * 3 * DII + i];
            part[4 + h] += g * fgw[h * 3 * DII + i];
        }
    }
    #pragma unroll
    for (int j = 0; j < 8; ++j) {
        float tot = blk_reduce_sum(part[j], sh, 256);
        if (tid == 0) {
            int h = j & 3;
            long long o = ((long long)b * NHH + h) * TT + t;
            if (j < 4) ig[o] = tot + igb[h];
            else       fg[o] = tot + fgb[h];
        }
        __syncthreads();
    }
}

__device__ __forceinline__ float log_sigmoidf(float x)
{
    return (x >= 0.f) ? -log1pf(expf(-x)) : (x - log1pf(expf(x)));
}

// inclusive cumsum of logsigmoid(fg) per (b,h); 1024-thread scan
__global__ void logsig_cumsum(const float* __restrict__ fg, float* __restrict__ cl)
{
    __shared__ float sc[TT];
    long long r = blockIdx.x;
    int t = threadIdx.x;
    sc[t] = log_sigmoidf(fg[r * TT + t]);
    __syncthreads();
    for (int off = 1; off < TT; off <<= 1) {
        float add = (t >= off) ? sc[t - off] : 0.f;
        __syncthreads();
        sc[t] += add;
        __syncthreads();
    }
    cl[r * TT + t] = sc[t];
}

// per-row decay weighting + stabilized normalization of qk (in place)
__global__ void att_row(float* __restrict__ att, const float* __restrict__ cl,
                        const float* __restrict__ ig)
{
    __shared__ float sh[256];
    int s = blockIdx.x;
    long long r = blockIdx.y;                         // b*NH + h
    float* row = att + (r * TT + s) * (long long)TT;
    const float* clr = cl + r * TT;
    const float* igr = ig + r * TT;
    float cls = clr[s];
    int tid = threadIdx.x;

    float mx = -INFINITY;
    for (int t = tid; t <= s; t += 256)
        mx = fmaxf(mx, cls - clr[t] + igr[t]);
    float maxD = blk_reduce_max(mx, sh, 256);

    float ssum = 0.f;
    for (int t = tid; t < TT; t += 256) {
        float cm = 0.f;
        if (t <= s) cm = row[t] * expf(cls - clr[t] + igr[t] - maxD);
        row[t] = cm;
        ssum += cm;
    }
    float tot = blk_reduce_sum(ssum, sh, 256);
    float inv = 1.f / (fmaxf(fabsf(tot), expf(-maxD)) + 1e-6f);
    for (int t = tid; t < TT; t += 256) row[t] *= inv;
}

// (hln + skip*xc) * silu(z)
__global__ void mlstm_combine(const float* __restrict__ hln, const float* __restrict__ xc,
                              const float* __restrict__ z, int ldz,
                              const float* __restrict__ skip,
                              float* __restrict__ out, int total)
{
    int idx = blockIdx.x * blockDim.x + threadIdx.x;
    if (idx >= total) return;
    int c = idx % DII;
    long long bt = idx / DII;
    float zz = z[bt * ldz + c];
    float sil = zz / (1.f + expf(-zz));
    out[idx] = (hln[idx] + skip[c] * xc[idx]) * sil;
}

__global__ void geglu(const float* __restrict__ up, int ld,
                      float* __restrict__ out, int total)
{
    int idx = blockIdx.x * blockDim.x + threadIdx.x;
    if (idx >= total) return;
    int c = idx % DUPP;
    long long bt = idx / DUPP;
    float g = up[bt * ld + c];
    float vv = up[bt * ld + DUPP + c];
    out[idx] = 0.5f * g * (1.f + erff(g * 0.70710678118f)) * vv;
}

__global__ void add_inplace(float* __restrict__ y, const float* __restrict__ x, int total)
{
    int idx = blockIdx.x * blockDim.x + threadIdx.x;
    if (idx < total) y[idx] += x[idx];
}

__global__ void transposeR(const float* __restrict__ R, float* __restrict__ Rt, int total)
{
    int idx = blockIdx.x * blockDim.x + threadIdx.x;   // over NH*4*128*128
    if (idx >= total) return;
    int o  = idx % DHS;
    int d  = (idx / DHS) % DHS;
    int gh = idx / (DHS * DHS);
    Rt[((long long)gh * DHS + d) * DHS + o] = R[((long long)gh * DHS + o) * DHS + d];
}

// ======================= sLSTM sequential scan =============================
// one 512-thread block per (b,h); thread = (gate g, channel o)
__global__ void slstm_scan(const float* __restrict__ gates,   // (4,B,S,D)
                           const float* __restrict__ Rt,      // (NH,4,d,o)
                           const float* __restrict__ bias,    // (4,NH,DHS)
                           float* __restrict__ y)              // (B,S,D)
{
    __shared__ float hs[DHS], cs[DHS], ns[DHS], ms[DHS];
    __shared__ float raws[4 * DHS];
    int b = blockIdx.x / NHH;
    int h = blockIdx.x % NHH;
    int tid = threadIdx.x;
    int g = tid >> 7;          // 0..3
    int o = tid & 127;
    if (tid < DHS) { hs[tid] = 0.f; cs[tid] = 0.f; ns[tid] = 0.f; ms[tid] = 0.f; }
    __syncthreads();

    const float* rp = Rt + ((long long)(h * 4 + g) * DHS) * DHS + o;  // + d*DHS
    const float bsv = bias[(g * NHH + h) * DHS + o];

    for (int t = 0; t < TT; ++t) {
        float rec = 0.f;
        #pragma unroll 4
        for (int d = 0; d < DHS; ++d)
            rec += hs[d] * rp[d * DHS];
        long long gidx = (((long long)g * BB + b) * TT + t) * DD + h * DHS + o;
        raws[g * DHS + o] = gates[gidx] + rec + bsv;
        __syncthreads();
        if (tid < DHS) {
            float ir = raws[0 * DHS + tid];
            float fr = raws[1 * DHS + tid];
            float zr = raws[2 * DHS + tid];
            float orr = raws[3 * DHS + tid];
            float logf_ = log_sigmoidf(fr);
            float mnew = fmaxf(ir, ms[tid] + logf_);
            float iv = expf(ir - mnew);
            float fv = expf(ms[tid] + logf_ - mnew);
            float cnew = fv * cs[tid] + iv * tanhf(zr);
            float nnew = fv * ns[tid] + iv;
            float hnew = (1.f / (1.f + expf(-orr))) * cnew / nnew;
            cs[tid] = cnew; ns[tid] = nnew; ms[tid] = mnew; hs[tid] = hnew;
            y[((long long)b * TT + t) * DD + h * DHS + tid] = hnew;
        }
        __syncthreads();
    }
}

// ======================= output mask =======================================
__global__ void apply_mask(float* __restrict__ out, const int* __restrict__ mask, int total)
{
    int idx = blockIdx.x * blockDim.x + threadIdx.x;
    if (idx >= total) return;
    long long bt = idx / CC;
    if (mask[bt] == 0) out[idx] = -10000.f;
}

// ======================= host orchestration ================================
static inline dim3 g1(long long n, int b) { return dim3((unsigned)((n + b - 1) / b)); }

extern "C" void kernel_launch(void* const* d_in, const int* in_sizes, int n_in,
                              void* d_out, int out_size, void* d_ws, size_t ws_size,
                              hipStream_t stream)
{
    (void)in_sizes; (void)n_in; (void)out_size; (void)ws_size;
    const float* x    = (const float*)d_in[0];
    const float* c1w  = (const float*)d_in[1];
    const float* c1b  = (const float*)d_in[2];
    const float* bn1g = (const float*)d_in[3];
    const float* bn1b = (const float*)d_in[4];
    const float* bn1m = (const float*)d_in[5];
    const float* bn1v = (const float*)d_in[6];
    const float* c2w  = (const float*)d_in[7];
    const float* c2b  = (const float*)d_in[8];
    const float* bn2g = (const float*)d_in[9];
    const float* bn2b = (const float*)d_in[10];
    const float* bn2m = (const float*)d_in[11];
    const float* bn2v = (const float*)d_in[12];
    const float* flnw = (const float*)d_in[13];
    const float* flnb = (const float*)d_in[14];
    const float* mlnw = (const float*)d_in[15];
    const float* mupw = (const float*)d_in[16];
    const float* mcw  = (const float*)d_in[17];
    const float* mcb  = (const float*)d_in[18];
    const float* mqw  = (const float*)d_in[19];
    const float* mkw  = (const float*)d_in[20];
    const float* mvw  = (const float*)d_in[21];
    const float* migw = (const float*)d_in[22];
    const float* migb = (const float*)d_in[23];
    const float* mfgw = (const float*)d_in[24];
    const float* mfgb = (const float*)d_in[25];
    const float* monw = (const float*)d_in[26];
    const float* mskip= (const float*)d_in[27];
    const float* mdw  = (const float*)d_in[28];
    const float* sln1w= (const float*)d_in[29];
    const float* scw  = (const float*)d_in[30];
    const float* scb  = (const float*)d_in[31];
    const float* siw  = (const float*)d_in[32];
    const float* sfw  = (const float*)d_in[33];
    const float* szw  = (const float*)d_in[34];
    const float* sow  = (const float*)d_in[35];
    const float* sR   = (const float*)d_in[36];
    const float* sb   = (const float*)d_in[37];
    const float* sgnw = (const float*)d_in[38];
    const float* sln2w= (const float*)d_in[39];
    const float* sffu = (const float*)d_in[40];
    const float* sffd = (const float*)d_in[41];
    const float* plnw = (const float*)d_in[42];
    const float* h1w  = (const float*)d_in[43];
    const float* h1b  = (const float*)d_in[44];
    const float* hlnw = (const float*)d_in[45];
    const float* hlnb = (const float*)d_in[46];
    const float* h2w  = (const float*)d_in[47];
    const float* h2b  = (const float*)d_in[48];
    const int*   amask= (const int*)d_in[49];

    float* ws = (float*)d_ws;
    float* buf_h   = ws;                                   // BT*512
    float* buf_ln  = buf_h   + (long long)BT * DD;         // BT*512
    float* buf_up  = buf_ln  + (long long)BT * DD;         // BT*2048
    float* buf_xc  = buf_up  + (long long)BT * 2 * DII;    // BT*1024
    float* buf_q   = buf_xc  + (long long)BT * DII;        // BT*1024
    float* buf_k   = buf_q   + (long long)BT * DII;        // BT*1024
    float* buf_v   = buf_k   + (long long)BT * DII;        // BT*1024
    float* buf_att = buf_v   + (long long)BT * DII;        // 16*1024*1024
    float* buf_ho  = buf_att + (long long)BB * NHH * TT * TT; // BT*1024
    float* buf_g   = buf_ho  + (long long)BT * DII;        // 4*BT*512
    float* buf_y   = buf_g   + 4LL * BT * DD;              // BT*512
    float* buf_Rt  = buf_y   + (long long)BT * DD;         // NH*4*128*128
    float* buf_ig  = buf_Rt  + (long long)NHH * 4 * DHS * DHS;
    float* buf_fg  = buf_ig  + (long long)BB * NHH * TT;
    float* buf_cl  = buf_fg  + (long long)BB * NHH * TT;

    float* out = (float*)d_out;

    // ---------------- front end -------------------------------------------
    conv_bn_relu<<<g1((long long)BT * 256, 256), 256, 0, stream>>>(
        x, FF, 256, c1w, c1b, bn1g, bn1b, bn1m, bn1v,
        buf_up, 1.f / (1.f + 1e-5f), BT * 256);
    conv_bn_relu<<<g1((long long)BT * DD, 256), 256, 0, stream>>>(
        buf_up, 256, DD, c2w, c2b, bn2g, bn2b, bn2m, bn2v,
        buf_xc, 1.f, BT * DD);
    ln_kernel<<<BT, 256, 0, stream>>>(buf_xc, flnw, flnb, buf_h, DD, 0);

    // ---------------- mLSTM block -----------------------------------------
    auto mblock = [&](int j) {
        const float* lnw_j  = mlnw  + (long long)j * DD;
        const float* upw_j  = mupw  + (long long)j * 2 * DII * DD;
        const float* cw_j   = mcw   + (long long)j * DII * 4;
        const float* cb_j   = mcb   + (long long)j * DII;
        const float* qw_j   = mqw   + (long long)j * NBB * 16;
        const float* kw_j   = mkw   + (long long)j * NBB * 16;
        const float* vw_j   = mvw   + (long long)j * NBB * 16;
        const float* igw_j  = migw  + (long long)j * NHH * 3 * DII;
        const float* igb_j  = migb  + (long long)j * NHH;
        const float* fgw_j  = mfgw  + (long long)j * NHH * 3 * DII;
        const float* fgb_j  = mfgb  + (long long)j * NHH;
        const float* onw_j  = monw  + (long long)j * DII;
        const float* skip_j = mskip + (long long)j * DII;
        const float* dw_j   = mdw   + (long long)j * DD * DII;

        ln_kernel<<<BT, 256, 0, stream>>>(buf_h, lnw_j, nullptr, buf_ln, DD, 0);

        // up projection: (BT,512) x (2048,512)^T
        gemm_bf16_wmma<<<dim3(2 * DII / 64, BT / 64, 1), 256, 0, stream>>>(
            buf_ln, upw_j, nullptr, buf_up, BT, 2 * DII, DD,
            DD, DD, 2 * DII, 1, 0, 0, 0, 0, 0, 0, 0, 1.f, 0, 0);

        // causal conv + silu on xm (first half of up)
        cconv_silu<<<g1((long long)BT * DII, 256), 256, 0, stream>>>(
            buf_up, 2 * DII, cw_j, cb_j, buf_xc, DII, DII, BT * DII);

        // headwise q/k from xc; v from xm
        headwise4<<<g1((long long)BT * NBB, 256), 256, 0, stream>>>(
            buf_xc, DII, qw_j, buf_q, DII, BT * NBB);
        headwise4<<<g1((long long)BT * NBB, 256), 256, 0, stream>>>(
            buf_xc, DII, kw_j, buf_k, DII, BT * NBB);
        headwise4<<<g1((long long)BT * NBB, 256), 256, 0, stream>>>(
            buf_up, 2 * DII, vw_j, buf_v, DII, BT * NBB);

        mlstm_gates<<<BT, 256, 0, stream>>>(buf_q, buf_k, buf_v,
                                            igw_j, igb_j, fgw_j, fgb_j,
                                            buf_ig, buf_fg);
        logsig_cumsum<<<BB * NHH, TT, 0, stream>>>(buf_fg, buf_cl);

        // QK^T per (b,h): M=N=S, K=DHM, alpha = 1/sqrt(DHM)
        gemm_bf16_wmma<<<dim3(TT / 64, TT / 64, BB * NHH), 256, 0, stream>>>(
            buf_q, buf_k, nullptr, buf_att, TT, TT, DHM,
            DII, DII, TT, NHH,
            (long long)TT * DII, DHM,
            (long long)TT * DII, DHM,
            (long long)NHH * TT * TT, (long long)TT * TT,
            0, 1.f / 16.f, 0, 0);

        att_row<<<dim3(TT, BB * NHH), 256, 0, stream>>>(buf_att, buf_cl, buf_ig);

        // (Cm/norm) @ V: W = v as (K=t, N=d)
        gemm_bf16_wmma<<<dim3(DHM / 64, TT / 64, BB * NHH), 256, 0, stream>>>(
            buf_att, buf_v, nullptr, buf_ho, TT, DHM, TT,
            TT, DII, DHM, NHH,
            (long long)NHH * TT * TT, (long long)TT * TT,
            (long long)TT * DII, DHM,
            (long long)NHH * TT * DHM, (long long)TT * DHM,
            1, 1.f, 0, 0);

        mh_ln_kernel<<<BB * NHH * TT, 256, 0, stream>>>(buf_ho, onw_j, buf_q);

        mlstm_combine<<<g1((long long)BT * DII, 256), 256, 0, stream>>>(
            buf_q, buf_xc, buf_up + DII, 2 * DII, skip_j, buf_k, BT * DII);

        // down projection, accumulate into residual stream
        gemm_bf16_wmma<<<dim3(DD / 64, BT / 64, 1), 256, 0, stream>>>(
            buf_k, dw_j, nullptr, buf_h, BT, DD, DII,
            DII, DII, DD, 1, 0, 0, 0, 0, 0, 0, 0, 1.f, 0, 1);
    };

    mblock(0);

    // ---------------- sLSTM block -----------------------------------------
    ln_kernel<<<BT, 256, 0, stream>>>(buf_h, sln1w, nullptr, buf_ln, DD, 0);
    cconv_silu<<<g1((long long)BT * DD, 256), 256, 0, stream>>>(
        buf_ln, DD, scw, scb, buf_xc, DD, DD, BT * DD);

    // per-head 128x128 gate projections (batched over heads via grid.z)
    const float* gate_src[4] = {buf_xc, buf_xc, buf_ln, buf_ln};
    const float* gate_w[4]   = {siw, sfw, szw, sow};
    for (int gi = 0; gi < 4; ++gi) {
        gemm_bf16_wmma<<<dim3(DHS / 64, BT / 64, NHH), 256, 0, stream>>>(
            gate_src[gi], gate_w[gi], nullptr, buf_g + (long long)gi * BT * DD,
            BT, DHS, DHS, DD, DHS, DD, 1,
            DHS, 0, (long long)DHS * DHS, 0, DHS, 0,
            0, 1.f, 0, 0);
    }

    transposeR<<<g1((long long)NHH * 4 * DHS * DHS, 256), 256, 0, stream>>>(
        sR, buf_Rt, NHH * 4 * DHS * DHS);
    slstm_scan<<<BB * NHH, 4 * DHS, 0, stream>>>(buf_g, buf_Rt, sb, buf_y);

    ln_grouped_kernel<<<BT * NHH, DHS, 0, stream>>>(buf_y, sgnw, buf_ln);
    add_inplace<<<g1((long long)BT * DD, 256), 256, 0, stream>>>(buf_h, buf_ln, BT * DD);

    // ---------------- FFN --------------------------------------------------
    ln_kernel<<<BT, 256, 0, stream>>>(buf_h, sln2w, nullptr, buf_ln, DD, 0);
    gemm_bf16_wmma<<<dim3(2 * DUPP / 64, BT / 64, 1), 256, 0, stream>>>(
        buf_ln, sffu, nullptr, buf_up, BT, 2 * DUPP, DD,
        DD, DD, 2 * DUPP, 1, 0, 0, 0, 0, 0, 0, 0, 1.f, 0, 0);
    geglu<<<g1((long long)BT * DUPP, 256), 256, 0, stream>>>(
        buf_up, 2 * DUPP, buf_xc, BT * DUPP);
    gemm_bf16_wmma<<<dim3(DD / 64, BT / 64, 1), 256, 0, stream>>>(
        buf_xc, sffd, nullptr, buf_h, BT, DD, DUPP,
        DUPP, DUPP, DD, 1, 0, 0, 0, 0, 0, 0, 0, 1.f, 0, 1);

    mblock(1);
    mblock(2);

    // ---------------- head -------------------------------------------------
    ln_kernel<<<BT, 256, 0, stream>>>(buf_h, plnw, nullptr, buf_ln, DD, 0);
    gemm_bf16_wmma<<<dim3(256 / 64, BT / 64, 1), 256, 0, stream>>>(
        buf_ln, h1w, h1b, buf_xc, BT, 256, DD,
        DD, DD, 256, 1, 0, 0, 0, 0, 0, 0, 0, 1.f, 0, 0);
    ln_kernel<<<BT, 256, 0, stream>>>(buf_xc, hlnw, hlnb, buf_k, 256, 1);
    gemm_bf16_wmma<<<dim3(CC / 64, BT / 64, 1), 256, 0, stream>>>(
        buf_k, h2w, h2b, out, BT, CC, 256,
        256, 256, CC, 1, 0, 0, 0, 0, 0, 0, 0, 1.f, 0, 0);
    apply_mask<<<g1((long long)BT * CC, 256), 256, 0, stream>>>(out, amask, BT * CC);
}
